// DISTS_SMIC_58626303590527
// MI455X (gfx1250) — compile-verified, matrix operand order checked by reading, not run on hardware
//
#include <hip/hip_runtime.h>
#include <hip/hip_fp16.h>
#include <math.h>

typedef __attribute__((ext_vector_type(16))) _Float16 v16h;
typedef __attribute__((ext_vector_type(8)))  _Float16 v8h;
typedef __attribute__((ext_vector_type(8)))  float    v8f;

#define C1_CONST 1e-6f
#define C2_CONST 1e-6f

// ---------------------------------------------------------------------------
// Weight repack: f32 [Co][Ci][3][3] -> f16 [9][Co][Cipad] (tap-major, zero pad)
// ---------------------------------------------------------------------------
__global__ void k_cvt_wf16(const float* __restrict__ w, _Float16* __restrict__ wh,
                           int Co, int Ci, int Cipad) {
  int idx = blockIdx.x * blockDim.x + threadIdx.x;
  if (idx >= 9 * Co * Cipad) return;
  int ci = idx % Cipad;
  int t2 = idx / Cipad;
  int co = t2 % Co;
  int t  = t2 / Co;
  float v = (ci < Ci) ? w[((size_t)co * Ci + ci) * 9 + t] : 0.0f;
  wh[idx] = (_Float16)v;
}

// ---------------------------------------------------------------------------
// NCHW f32 image (3 channels) -> NHWC f16 padded to 32 channels
// ---------------------------------------------------------------------------
__global__ void k_img2nhwc(const float* __restrict__ src, _Float16* __restrict__ dst,
                           int HW) {
  int idx = blockIdx.x * blockDim.x + threadIdx.x;
  if (idx >= HW * 32) return;
  int n = idx >> 5, c = idx & 31;
  float v = (c < 3) ? src[(size_t)c * HW + n] : 0.0f;
  dst[idx] = (_Float16)v;
}

// ---------------------------------------------------------------------------
// 3x3 conv (pad=1) as 9 shifted GEMMs on v_wmma_f32_16x16x32_f16.
// in : NHWC f16 [H][W][Ci]   (Ci multiple of 32)
// wh : f16 [9][Co][Ci]
// out: NHWC f16 [H][W][Co], bias + ReLU fused.
// Block = 128 threads = 4 wave32. Each wave computes 64 Co x 16 pixels:
// per K-chunk it loads B once and 4 A fragments, issuing 4 independent WMMAs
// (same B, different acc -> no RAW hazard NOPs, 32768 MACs per 5120B loaded).
// Out-of-bounds taps read from a zeroed page -> branch-free inner loop, no
// EXEC-mask manipulation around the WMMA.
// ---------------------------------------------------------------------------
__global__ __launch_bounds__(128)
void k_conv3x3_wmma(const _Float16* __restrict__ in, const _Float16* __restrict__ wh,
                    const float* __restrict__ bias, _Float16* __restrict__ out,
                    const _Float16* __restrict__ zpage,
                    int H, int W, int Ci, int Co) {
  const int lane = threadIdx.x & 31;
  const int wave = threadIdx.x >> 5;
  const int n0   = (blockIdx.x * 4 + wave) * 16;   // W % 16 == 0 -> row aligned
  const int co0  = blockIdx.y * 64;
  const int col  = lane & 15;                      // A row (M) / B,C column (N)
  const int kb   = (lane < 16) ? 0 : 8;            // per-half-wave K sub-offset
  const int h    = n0 / W;
  const int w    = (n0 - h * W) + col;
  const int npix = n0 + col;

  v8f acc[4] = {{}, {}, {}, {}};
  const size_t tstride = (size_t)Co * Ci;
  const _Float16* abase = wh + (size_t)(co0 + col) * Ci + kb;

#pragma unroll
  for (int t = 0; t < 9; ++t) {
    const int r = t / 3, s = t % 3;
    const int hy = h + r - 1;
    const int wx = w + s - 1;
    const bool valid = ((unsigned)hy < (unsigned)H) && ((unsigned)wx < (unsigned)W);
    const _Float16* bsrc =
        valid ? (in + ((size_t)hy * W + wx) * Ci + kb) : (zpage + kb);
    const _Float16* arow = abase + (size_t)t * tstride;
    __builtin_prefetch(bsrc, 0, 1);

    for (int c0 = 0; c0 < Ci; c0 += 32) {
      // B fragment (shared across the 4 Co sub-tiles)
      v8h blo = *(const v8h*)(bsrc + c0);
      v8h bhi = *(const v8h*)(bsrc + c0 + 16);
      v16h b;
#pragma unroll
      for (int j = 0; j < 8; ++j) { b[j] = blo[j]; b[j + 8] = bhi[j]; }

#pragma unroll
      for (int u = 0; u < 4; ++u) {
        const _Float16* ar = arow + (size_t)u * 16 * Ci + c0;
        v8h alo = *(const v8h*)(ar);
        v8h ahi = *(const v8h*)(ar + 16);
        v16h a;
#pragma unroll
        for (int j = 0; j < 8; ++j) { a[j] = alo[j]; a[j + 8] = ahi[j]; }
        acc[u] = __builtin_amdgcn_wmma_f32_16x16x32_f16(false, a, false, b,
                                                        (short)0, acc[u],
                                                        false, false);
      }
    }
  }

  // Writeback: 4 sub-tiles x 8 contiguous output channels -> 16B stores (NHWC)
  const int mbase = (lane < 16) ? 0 : 8;
  _Float16* obase = out + (size_t)npix * Co + co0 + mbase;
#pragma unroll
  for (int u = 0; u < 4; ++u) {
    v8h ov;
#pragma unroll
    for (int rr = 0; rr < 8; ++rr) {
      float v = acc[u][rr] + bias[co0 + u * 16 + mbase + rr];
      ov[rr] = (_Float16)fmaxf(v, 0.0f);
    }
    *(v8h*)(obase + u * 16) = ov;
  }
}

// ---------------------------------------------------------------------------
// l2pool (NHWC f16): sqrt( conv(x^2, hanning3x3, stride 2, pad 1, dw) + 1e-12 )
// ---------------------------------------------------------------------------
__global__ void k_l2pool(const _Float16* __restrict__ in, _Float16* __restrict__ out,
                         int C, int H, int W) {
  const int Ho = H / 2, Wo = W / 2;
  int idx = blockIdx.x * blockDim.x + threadIdx.x;
  if (idx >= Ho * Wo * C) return;
  int c = idx % C;
  int n = idx / C;
  int wo = n % Wo;
  int ho = n / Wo;
  const float g[3] = {0.5f, 1.0f, 0.5f};
  float s = 0.0f;
#pragma unroll
  for (int r = 0; r < 3; ++r) {
    int hy = 2 * ho + r - 1;
    if ((unsigned)hy >= (unsigned)H) continue;
#pragma unroll
    for (int q = 0; q < 3; ++q) {
      int wx = 2 * wo + q - 1;
      if ((unsigned)wx >= (unsigned)W) continue;
      float v = (float)in[((size_t)hy * W + wx) * C + c];
      s += g[r] * g[q] * v * v;
    }
  }
  out[idx] = (_Float16)sqrtf(s * 0.25f + 1e-12f);
}

// ---------------------------------------------------------------------------
// Helpers: init, zero, w_sum reduce, final combine
// ---------------------------------------------------------------------------
__global__ void k_init(float* a) {
  if (threadIdx.x == 0) { a[0] = 0.f; a[1] = 0.f; a[2] = 0.f; a[3] = 0.f; }
}

__global__ void k_zero(float* p, int n) {
  int i = blockIdx.x * blockDim.x + threadIdx.x;
  if (i < n) p[i] = 0.f;
}

__global__ void k_wsum(const float* __restrict__ a, const float* __restrict__ b,
                       int n, float* ws) {
  __shared__ float sh[256];
  float s = 0.f;
  for (int i = threadIdx.x; i < n; i += 256) s += a[i] + b[i];
  sh[threadIdx.x] = s; __syncthreads();
  for (int off = 128; off > 0; off >>= 1) {
    if (threadIdx.x < off) sh[threadIdx.x] += sh[threadIdx.x + off];
    __syncthreads();
  }
  if (threadIdx.x == 0) ws[0] = sh[0];
}

__global__ void k_final(const float* acc, float* out) {
  if (threadIdx.x == 0) out[0] = 1.0f - (acc[0] + acc[1]) + acc[2];
}

// ---------------------------------------------------------------------------
// Stage stats, k=0 (f32 NCHW originals): one block per channel.
// ---------------------------------------------------------------------------
__global__ void k_stats_nchw(const float* __restrict__ fx, const float* __restrict__ fy,
                             const float* __restrict__ al, const float* __restrict__ be,
                             const float* __restrict__ wsum, int HW, float* acc) {
  __shared__ float sh[256];
  int c = blockIdx.x;
  const float* px = fx + (size_t)c * HW;
  const float* py = fy + (size_t)c * HW;
  float sx = 0, sy = 0, sxx = 0, syy = 0, sxy = 0;
  for (int i = threadIdx.x; i < HW; i += 256) {
    float a = px[i], b = py[i];
    sx += a; sy += b; sxx += a * a; syy += b * b; sxy += a * b;
  }
  float sums[5] = {sx, sy, sxx, syy, sxy};
  float tot[5];
  for (int v = 0; v < 5; ++v) {
    sh[threadIdx.x] = sums[v]; __syncthreads();
    for (int off = 128; off > 0; off >>= 1) {
      if (threadIdx.x < off) sh[threadIdx.x] += sh[threadIdx.x + off];
      __syncthreads();
    }
    tot[v] = sh[0]; __syncthreads();
  }
  if (threadIdx.x == 0) {
    float inv = 1.0f / (float)HW;
    float xm = tot[0] * inv, ym = tot[1] * inv;
    float S1 = (2.f * xm * ym + C1_CONST) / (xm * xm + ym * ym + C1_CONST);
    float xv = tot[2] * inv - xm * xm;
    float yv = tot[3] * inv - ym * ym;
    float cv = tot[4] * inv - xm * ym;
    float S2 = (2.f * cv + C2_CONST) / (xv + yv + C2_CONST);
    float iw = 1.0f / wsum[0];
    atomicAdd(&acc[0], al[c] * iw * S1);
    atomicAdd(&acc[1], be[c] * iw * S2);
  }
}

// ---------------------------------------------------------------------------
// Stage stats, NHWC f16 features: coalesced per-channel moment accumulation
// (thread = channel, gridDim.y = spatial slices), then finalize.
// ---------------------------------------------------------------------------
__global__ void k_moments(const _Float16* __restrict__ fx, const _Float16* __restrict__ fy,
                          int C, int HW, float* __restrict__ mom) {
  int c = blockIdx.x * blockDim.x + threadIdx.x;
  if (c >= C) return;
  int i0 = (int)(((long long)HW * blockIdx.y) / gridDim.y);
  int i1 = (int)(((long long)HW * (blockIdx.y + 1)) / gridDim.y);
  float sx = 0, sy = 0, sxx = 0, syy = 0, sxy = 0;
  for (int i = i0; i < i1; ++i) {
    float a = (float)fx[(size_t)i * C + c];
    float b = (float)fy[(size_t)i * C + c];
    sx += a; sy += b; sxx += a * a; syy += b * b; sxy += a * b;
  }
  atomicAdd(&mom[c * 5 + 0], sx);
  atomicAdd(&mom[c * 5 + 1], sy);
  atomicAdd(&mom[c * 5 + 2], sxx);
  atomicAdd(&mom[c * 5 + 3], syy);
  atomicAdd(&mom[c * 5 + 4], sxy);
}

__global__ void k_stats_fin(const float* __restrict__ mom,
                            const float* __restrict__ al, const float* __restrict__ be,
                            const float* __restrict__ wsum, int HW, int C, float* acc) {
  int c = blockIdx.x * blockDim.x + threadIdx.x;
  if (c >= C) return;
  float inv = 1.0f / (float)HW;
  float xm = mom[c * 5 + 0] * inv, ym = mom[c * 5 + 1] * inv;
  float S1 = (2.f * xm * ym + C1_CONST) / (xm * xm + ym * ym + C1_CONST);
  float xv = mom[c * 5 + 2] * inv - xm * xm;
  float yv = mom[c * 5 + 3] * inv - ym * ym;
  float cv = mom[c * 5 + 4] * inv - xm * ym;
  float S2 = (2.f * cv + C2_CONST) / (xv + yv + C2_CONST);
  float iw = 1.0f / wsum[0];
  atomicAdd(&acc[0], al[c] * iw * S1);
  atomicAdd(&acc[1], be[c] * iw * S2);
}

// ---------------------------------------------------------------------------
// 1x1 projection (NHWC): mic[n][o] = sum_c pj[o][c] * f[n][c]
// ---------------------------------------------------------------------------
__global__ void k_proj(const float* __restrict__ pj, const _Float16* __restrict__ f,
                       float* __restrict__ out, int C, int HW) {
  int idx = blockIdx.x * blockDim.x + threadIdx.x;
  if (idx >= HW * 32) return;
  int n = idx >> 5;
  int o = idx & 31;
  const float* pr = pj + (size_t)o * C;
  const _Float16* fr = f + (size_t)n * C;
  float s = 0.f;
  for (int c = 0; c < C; ++c) s += pr[c] * (float)fr[c];
  out[idx] = s;
}

// ---------------------------------------------------------------------------
// Per-patch SMIC score (k in {3,4}); NHWC inputs. One block per patch.
// ---------------------------------------------------------------------------
__global__ void k_patch(const float* __restrict__ mic0, const float* __restrict__ mic1,
                        const _Float16* __restrict__ f0, const _Float16* __restrict__ f1,
                        int C, int H, int W, int Hp, int Wp,
                        float invP, float* score) {
  __shared__ float sh[256];
  __shared__ float micchn;
  const int p  = blockIdx.x;
  const int ph = (p / Wp) * 6;
  const int pw = (p % Wp) * 6;
  const float inv49 = 1.0f / 49.0f;

  // Phase A: r^2 correlation over the 32 mic channels (mic is [n][32] f32)
  float r2 = 0.0f;
  if (threadIdx.x < 32) {
    float s0 = 0, s1 = 0, s00 = 0, s11 = 0, s01 = 0;
    for (int i = 0; i < 7; ++i)
      for (int j = 0; j < 7; ++j) {
        size_t n = (size_t)(ph + i) * W + (pw + j);
        float a = mic0[n * 32 + threadIdx.x];
        float b = mic1[n * 32 + threadIdx.x];
        s0 += a; s1 += b; s00 += a * a; s11 += b * b; s01 += a * b;
      }
    float m0 = s0 * inv49, m1 = s1 * inv49;
    float cov = s01 * inv49 - m0 * m1;
    float v0  = s00 * inv49 - m0 * m0;
    float v1  = s11 * inv49 - m1 * m1;
    r2 = cov * cov / (v0 * v1 + 1e-12f);
  }
  sh[threadIdx.x] = (threadIdx.x < 32) ? r2 : 0.0f;
  __syncthreads();
  if (threadIdx.x == 0) {
    float s = 0.f;
    for (int i = 0; i < 32; ++i) s += sh[i];
    micchn = s * (1.0f / 32.0f);
  }
  __syncthreads();

  // Phase B: S1p / S2p over feature channels (coalesced over c = tid)
  float s1sum = 0.f, s2sum = 0.f;
  for (int c = threadIdx.x; c < C; c += 256) {
    float sx = 0, sy = 0, sxx = 0, syy = 0, sxy = 0;
    for (int i = 0; i < 7; ++i)
      for (int j = 0; j < 7; ++j) {
        size_t n = (size_t)(ph + i) * W + (pw + j);
        float a = (float)f0[n * C + c];
        float b = (float)f1[n * C + c];
        sx += a; sy += b; sxx += a * a; syy += b * b; sxy += a * b;
      }
    float xm = sx * inv49, ym = sy * inv49;
    s1sum += (2.f * xm * ym + C1_CONST) / (xm * xm + ym * ym + C1_CONST);
    float xv = sxx * inv49 - xm * xm;
    float yv = syy * inv49 - ym * ym;
    float cv = sxy * inv49 - xm * ym;
    s2sum += (2.f * cv + C2_CONST) / (xv + yv + C2_CONST);
  }
  sh[threadIdx.x] = s1sum; __syncthreads();
  for (int off = 128; off > 0; off >>= 1) {
    if (threadIdx.x < off) sh[threadIdx.x] += sh[threadIdx.x + off];
    __syncthreads();
  }
  float S1p = sh[0] / (float)C; __syncthreads();
  sh[threadIdx.x] = s2sum; __syncthreads();
  for (int off = 128; off > 0; off >>= 1) {
    if (threadIdx.x < off) sh[threadIdx.x] += sh[threadIdx.x + off];
    __syncthreads();
  }
  if (threadIdx.x == 0) {
    float S2p = sh[0] / (float)C;
    float per = (1.0f - micchn) * (2.0f - S1p - S2p);
    atomicAdd(score, per * invP);
  }
}

// ---------------------------------------------------------------------------
// Host orchestration
// ---------------------------------------------------------------------------
extern "C" void kernel_launch(void* const* d_in, const int* in_sizes, int n_in,
                              void* d_out, int out_size, void* d_ws, size_t ws_size,
                              hipStream_t stream) {
  (void)in_sizes; (void)out_size; (void)ws_size;
  if (n_in < 32) return;

  const float* x     = (const float*)d_in[0];
  const float* y     = (const float*)d_in[1];
  const float* alpha = (const float*)d_in[2];
  const float* beta  = (const float*)d_in[3];
  const float* vw[13]; const float* vb[13];
  for (int i = 0; i < 13; ++i) {
    vw[i] = (const float*)d_in[4 + i];
    vb[i] = (const float*)d_in[17 + i];
  }
  const float* pj1 = (const float*)d_in[30];
  const float* pj2 = (const float*)d_in[31];
  float* out = (float*)d_out;

  static const int CO[13]  = {64, 64, 128, 128, 256, 256, 256, 512, 512, 512, 512, 512, 512};
  static const int CI[13]  = {3, 64, 64, 128, 128, 256, 256, 256, 512, 512, 512, 512, 512};
  static const int CIP[13] = {32, 64, 64, 128, 128, 256, 256, 256, 512, 512, 512, 512, 512};

  char* wp = (char*)d_ws;
  auto alloc = [&](size_t bytes) -> void* {
    void* p = (void*)wp;
    wp += (bytes + 255) & ~(size_t)255;
    return p;
  };

  _Float16* wh[13];
  for (int l = 0; l < 13; ++l)
    wh[l] = (_Float16*)alloc((size_t)9 * CO[l] * CIP[l] * sizeof(_Float16));

  const size_t HW0 = 512 * 512;
  _Float16* xh = (_Float16*)alloc(HW0 * 32 * sizeof(_Float16));
  _Float16* yh = (_Float16*)alloc(HW0 * 32 * sizeof(_Float16));

  // feats[im][s] f16 NHWC: 64@512, 128@256, 256@128, 512@64, 512@32
  const size_t fs[5] = {HW0 * 64, (size_t)256 * 256 * 128, (size_t)128 * 128 * 256,
                        (size_t)64 * 64 * 512, (size_t)32 * 32 * 512};
  _Float16* F[2][5];
  for (int im = 0; im < 2; ++im)
    for (int s = 0; s < 5; ++s) F[im][s] = (_Float16*)alloc(fs[s] * sizeof(_Float16));

  _Float16* tmpA = (_Float16*)alloc(HW0 * 64 * sizeof(_Float16));
  _Float16* tmpB = (_Float16*)alloc((size_t)256 * 256 * 128 * sizeof(_Float16));
  _Float16* zpage = (_Float16*)alloc(512 * sizeof(_Float16));   // zeroed border page
  float* mic0 = (float*)alloc((size_t)32 * 128 * 128 * sizeof(float));
  float* mic1 = (float*)alloc((size_t)32 * 128 * 128 * sizeof(float));
  float* mom  = (float*)alloc((size_t)512 * 5 * sizeof(float));
  float* accum = (float*)alloc(8 * sizeof(float)); // [0]=dist1 [1]=dist2 [2]=score34 [3]=wsum

  k_init<<<1, 32, 0, stream>>>(accum);
  k_zero<<<1, 256, 0, stream>>>((float*)zpage, 256);   // 512 halves = 256 f32
  k_wsum<<<1, 256, 0, stream>>>(alpha, beta, 1475, accum + 3);

  for (int l = 0; l < 13; ++l) {
    int total = 9 * CO[l] * CIP[l];
    k_cvt_wf16<<<(total + 255) / 256, 256, 0, stream>>>(vw[l], wh[l], CO[l], CI[l], CIP[l]);
  }

  auto conv = [&](const _Float16* src, int l, _Float16* dst, int Hc) {
    int HWc = Hc * Hc;
    dim3 grid(HWc / 64, CO[l] / 64);
    k_conv3x3_wmma<<<grid, 128, 0, stream>>>(src, wh[l], vb[l], dst, zpage, Hc, Hc,
                                             CIP[l], CO[l]);
  };
  auto pool = [&](const _Float16* src, _Float16* dst, int C, int Hc) {
    int total = (Hc / 2) * (Hc / 2) * C;
    k_l2pool<<<(total + 255) / 256, 256, 0, stream>>>(src, dst, C, Hc, Hc);
  };

  k_img2nhwc<<<(int)((HW0 * 32 + 255) / 256), 256, 0, stream>>>(x, xh, (int)HW0);
  k_img2nhwc<<<(int)((HW0 * 32 + 255) / 256), 256, 0, stream>>>(y, yh, (int)HW0);

  for (int im = 0; im < 2; ++im) {
    const _Float16* img = im ? yh : xh;
    _Float16** Fi = F[im];
    conv(img, 0, tmpA, 512);
    conv(tmpA, 1, Fi[0], 512);
    pool(Fi[0], tmpA, 64, 512);
    conv(tmpA, 2, tmpB, 256);
    conv(tmpB, 3, Fi[1], 256);
    pool(Fi[1], tmpA, 128, 256);
    conv(tmpA, 4, tmpB, 128);
    conv(tmpB, 5, tmpA, 128);
    conv(tmpA, 6, Fi[2], 128);
    pool(Fi[2], tmpA, 256, 128);
    conv(tmpA, 7, tmpB, 64);
    conv(tmpB, 8, tmpA, 64);
    conv(tmpA, 9, Fi[3], 64);
    pool(Fi[3], tmpA, 512, 64);
    conv(tmpA, 10, tmpB, 32);
    conv(tmpB, 11, tmpA, 32);
    conv(tmpA, 12, Fi[4], 32);
  }

  // k = 0: original f32 NCHW inputs
  k_stats_nchw<<<3, 256, 0, stream>>>(x, y, alpha + 0, beta + 0, accum + 3,
                                      (int)HW0, accum);
  // k = 1, 2, 5: NHWC f16 features via coalesced moments
  struct { int fidx, C, HW, off, slices; } st[3] = {
    {0, 64, 512 * 512, 3, 64},
    {1, 128, 256 * 256, 67, 16},
    {4, 512, 32 * 32, 963, 4},
  };
  for (int q = 0; q < 3; ++q) {
    k_zero<<<(st[q].C * 5 + 255) / 256, 256, 0, stream>>>(mom, st[q].C * 5);
    dim3 mg((st[q].C + 255) / 256, st[q].slices);
    k_moments<<<mg, 256, 0, stream>>>(F[0][st[q].fidx], F[1][st[q].fidx],
                                      st[q].C, st[q].HW, mom);
    k_stats_fin<<<(st[q].C + 255) / 256, 256, 0, stream>>>(
        mom, alpha + st[q].off, beta + st[q].off, accum + 3, st[q].HW, st[q].C, accum);
  }

  // k = 3: feats[3] = 256 @ 128x128
  {
    int Hc = 128, C = 256, HWc = Hc * Hc;
    k_proj<<<(32 * HWc + 255) / 256, 256, 0, stream>>>(pj1, F[0][2], mic0, C, HWc);
    k_proj<<<(32 * HWc + 255) / 256, 256, 0, stream>>>(pj1, F[1][2], mic1, C, HWc);
    int Hp = (Hc - 7) / 6 + 1;        // 21
    int P = Hp * Hp;                  // 441
    k_patch<<<P, 256, 0, stream>>>(mic0, mic1, F[0][2], F[1][2], C, Hc, Hc, Hp, Hp,
                                   1.0f / (float)P, accum + 2);
  }
  // k = 4: feats[4] = 512 @ 64x64
  {
    int Hc = 64, C = 512, HWc = Hc * Hc;
    k_proj<<<(32 * HWc + 255) / 256, 256, 0, stream>>>(pj2, F[0][3], mic0, C, HWc);
    k_proj<<<(32 * HWc + 255) / 256, 256, 0, stream>>>(pj2, F[1][3], mic1, C, HWc);
    int Hp = (Hc - 7) / 6 + 1;        // 10
    int P = Hp * Hp;                  // 100
    k_patch<<<P, 256, 0, stream>>>(mic0, mic1, F[0][3], F[1][3], C, Hc, Hc, Hp, Hp,
                                   1.0f / (float)P, accum + 2);
  }

  k_final<<<1, 32, 0, stream>>>(accum, out);
}
